// CARAFE_31181462569349
// MI455X (gfx1250) — matrix operand order repaired and checked by reading, not compiled, not run on hardware
//
#include <hip/hip_runtime.h>
#include <hip/hip_bf16.h>

typedef __attribute__((ext_vector_type(16))) __bf16 v16bf;
typedef __attribute__((ext_vector_type(8)))  float  v8f;

union FragU { uint4 u[2]; v16bf v; };

#define CIN   32
#define MID   64
#define ENC   1000
#define MPAD  1008      // out2 row stride (floats)
#define KTOT  1728      // 27 taps * 64 ch
#define NSPAT 32768     // 32^3
#define NTOT  65536     // 2 * 32^3

// ---------------- Kernel A: 1x1x1 conv + BN1 + ReLU -> channels-last bf16 hi/lo
__global__ __launch_bounds__(256) void k_pointwise(
        const float* __restrict__ X,  const float* __restrict__ wc,
        const float* __restrict__ g1, const float* __restrict__ b1,
        const float* __restrict__ m1, const float* __restrict__ v1,
        __hip_bfloat16* __restrict__ W1hi, __hip_bfloat16* __restrict__ W1lo)
{
    int pos = blockIdx.x * 256 + threadIdx.x;        // 0..65535
    int b = pos >> 15, r = pos & 32767;
    const float* xp = X + (size_t)b * CIN * NSPAT + r;
    float xin[CIN];
    #pragma unroll
    for (int ci = 0; ci < CIN; ++ci) xin[ci] = xp[(size_t)ci * NSPAT];
    __hip_bfloat16* oh = W1hi + (size_t)pos * MID;
    __hip_bfloat16* ol = W1lo + (size_t)pos * MID;
    for (int co = 0; co < MID; ++co) {
        float acc = 0.f;
        const float* w = wc + co * CIN;
        #pragma unroll
        for (int ci = 0; ci < CIN; ++ci) acc = fmaf(xin[ci], w[ci], acc);
        float inv = g1[co] * rsqrtf(v1[co] + 1e-5f);
        float val = fmaxf(acc * inv + (b1[co] - m1[co] * inv), 0.f);
        __hip_bfloat16 h = __float2bfloat16(val);
        oh[co] = h;
        ol[co] = __float2bfloat16(val - __bfloat162float(h));
    }
}

// ---------------- Kernel P: repack w_enc -> bf16 hi/lo A[1024][k=tap*64+ci] --
__global__ __launch_bounds__(256) void k_pack_w(
        const float* __restrict__ we,
        __hip_bfloat16* __restrict__ Aphi, __hip_bfloat16* __restrict__ Aplo)
{
    int idx = blockIdx.x * 256 + threadIdx.x;        // < 1024*1728
    int m = idx / KTOT, k = idx - m * KTOT;
    int tap = k >> 6, ci = k & 63;
    float v = 0.f;
    if (m < ENC) v = we[(size_t)m * KTOT + ci * 27 + tap];
    __hip_bfloat16 h = __float2bfloat16(v);
    Aphi[(size_t)m * KTOT + k] = h;
    Aplo[(size_t)m * KTOT + k] = __float2bfloat16(v - __bfloat162float(h));
}

// ---------------- Kernel B: 3x3x3 conv as bf16x3 WMMA GEMM + BN2 -------------
// C[1008][65536] = A[1024][1728] x B[1728][65536], B = implicit im2col of W1.
// A = Ahi+Alo, B = Bhi+Blo; C ~= Ahi*Bhi + Alo*Bhi + Ahi*Blo  (fp32-quality)
__global__ __launch_bounds__(256) void k_conv_gemm(
        const __hip_bfloat16* __restrict__ Aphi,
        const __hip_bfloat16* __restrict__ Aplo,
        const __hip_bfloat16* __restrict__ W1hi,
        const __hip_bfloat16* __restrict__ W1lo,
        const float* __restrict__ g2, const float* __restrict__ b2,
        const float* __restrict__ m2, const float* __restrict__ v2,
        float* __restrict__ out2)
{
    __shared__ uint4 sAhi[512], sAlo[512];   // 128 rows x 32 bf16 each
    __shared__ uint4 sBhi[256], sBlo[256];   // 64 n   x 32 bf16 each (n-major)
    __shared__ float sInv[128], sBias[128];
    __shared__ float sC[64 * 132];           // n-major f32 result, padded stride

    const int t    = threadIdx.x;
    const int lane = t & 31, wv = t >> 5;
    const int lm   = lane & 15, hi = lane >> 4;
    const int n0   = blockIdx.x * 64;
    const int m0   = blockIdx.y * 128;

    if (t < 128) {
        int m = m0 + t;
        float inv = 0.f, bias = 0.f;
        if (m < ENC) {
            inv  = g2[m] * rsqrtf(v2[m] + 1e-5f);
            bias = b2[m] - m2[m] * inv;
        }
        sInv[t] = inv; sBias[t] = bias;
    }

    // per-thread B-staging coordinates (fixed n, 8 channels per thread)
    const int dn = t >> 2, gq = t & 3;
    const int nb = n0 + dn;
    const int bb = nb >> 15, rr = nb & 32767;
    const int z = rr >> 10, y = (rr >> 5) & 31, x = rr & 31;

    const uint4* Ah4 = (const uint4*)Aphi;   // 216 uint4 per A row
    const uint4* Al4 = (const uint4*)Aplo;
    const uint4* Wh4 = (const uint4*)W1hi;   // 8 uint4 per spatial point
    const uint4* Wl4 = (const uint4*)W1lo;
    const int arow = t >> 2, aq = t & 3;

    v8f acc[4];
    #pragma unroll
    for (int j = 0; j < 4; ++j)
        #pragma unroll
        for (int e = 0; e < 8; ++e) acc[j][e] = 0.f;

    for (int kt = 0; kt < 54; ++kt) {           // K = 54 * 32
        int tap = kt >> 1;
        int dz = tap / 9, rt = tap - dz * 9, dy = rt / 3, dx = rt - dy * 3;

        size_t a0 = (size_t)(m0 + arow)      * 216 + kt * 4 + aq;
        size_t a1 = (size_t)(m0 + 64 + arow) * 216 + kt * 4 + aq;
        sAhi[t]       = Ah4[a0];
        sAhi[t + 256] = Ah4[a1];
        sAlo[t]       = Al4[a0];
        sAlo[t + 256] = Al4[a1];

        int zz = z + dz - 1, yy = y + dy - 1, xx = x + dx - 1;
        uint4 bh = make_uint4(0u, 0u, 0u, 0u);
        uint4 bl = make_uint4(0u, 0u, 0u, 0u);
        if (((unsigned)zz < 32u) & ((unsigned)yy < 32u) & ((unsigned)xx < 32u)) {
            size_t spat = (((size_t)bb * 32 + zz) * 32 + yy) * 32 + xx;
            size_t gi = spat * 8 + (kt & 1) * 4 + gq;
            bh = Wh4[gi];
            bl = Wl4[gi];
        }
        sBhi[t] = bh;
        sBlo[t] = bl;
        __syncthreads();

        // A fragments: lane = {m = lm, khalf = hi}; K chunks [8h,8h+8) & [16+8h,..)
        FragU ah, al;
        int ar = (wv * 16 + lm) * 4;
        ah.u[0] = sAhi[ar + hi];     ah.u[1] = sAhi[ar + 2 + hi];
        al.u[0] = sAlo[ar + hi];     al.u[1] = sAlo[ar + 2 + hi];
        #pragma unroll
        for (int j = 0; j < 4; ++j) {
            // B fragments: lane = {n = lm, khalf = hi}; K chunks [8h,8h+8) & [16+8h,..), n-major
            FragU bhf, blf;
            int br = (j * 16 + lm) * 4 + hi;
            bhf.u[0] = sBhi[br];     bhf.u[1] = sBhi[br + 2];
            blf.u[0] = sBlo[br];     blf.u[1] = sBlo[br + 2];
            acc[j] = __builtin_amdgcn_wmma_f32_16x16x32_bf16(
                false, ah.v, false, bhf.v, (short)0, acc[j], false, false);
            acc[j] = __builtin_amdgcn_wmma_f32_16x16x32_bf16(
                false, al.v, false, bhf.v, (short)0, acc[j], false, false);
            acc[j] = __builtin_amdgcn_wmma_f32_16x16x32_bf16(
                false, ah.v, false, blf.v, (short)0, acc[j], false, false);
        }
        __syncthreads();
    }

    // BN2 + transpose through LDS for coalesced [n][m] stores
    #pragma unroll
    for (int j = 0; j < 4; ++j) {
        #pragma unroll
        for (int rg = 0; rg < 8; ++rg) {
            int ml = wv * 16 + rg + hi * 8;       // C layout: VGPR r -> M=r / M=8+r
            int nl = j * 16 + lm;
            sC[nl * 132 + ml] = acc[j][rg] * sInv[ml] + sBias[ml];
        }
    }
    __syncthreads();

    int nl = t >> 2, ch = t & 3;
    size_t rowbase = (size_t)(n0 + nl) * MPAD + m0 + ch * 32;
    #pragma unroll
    for (int f = 0; f < 8; ++f) {
        int mg = m0 + ch * 32 + f * 4;
        if (mg < MPAD) {
            float4 vv = *(const float4*)&sC[nl * 132 + ch * 32 + f * 4];
            *(float4*)(out2 + rowbase + f * 4) = vv;
        }
    }
}

// ---------------- Kernel C: pixel-shuffle + softmax(125) + reassembly --------
__global__ __launch_bounds__(256) void k_carafe(
        const float* __restrict__ out2, const float* __restrict__ X,
        float* __restrict__ out)
{
    __shared__ float sLog[1000];
    __shared__ float sX[32 * 125];
    __shared__ float sW[8 * 125];

    const int t  = threadIdx.x;
    const int nc = blockIdx.x;                       // coarse cell
    const int b  = nc >> 15, rr = nc & 32767;
    const int h0 = rr >> 10, w0 = (rr >> 5) & 31, d0 = rr & 31;

    const float* lg = out2 + (size_t)nc * MPAD;
    for (int i = t; i < 1000; i += 256) sLog[i] = lg[i];

    for (int idx = t; idx < 32 * 125; idx += 256) {
        int c = idx / 125, tap = idx - c * 125;
        int i = tap / 25, r2 = tap - i * 25, j = r2 / 5, l = r2 - j * 5;
        int xh = h0 + i - 2, yw = w0 + j - 2, zd = d0 + l - 2;
        float v = 0.f;
        if (((unsigned)xh < 32u) & ((unsigned)yw < 32u) & ((unsigned)zd < 32u))
            v = X[(((size_t)(b * 32 + c) * 32 + xh) * 32 + yw) * 32 + zd];
        sX[idx] = v;
    }
    __syncthreads();

    {   // one wave32 per sub-position softmax (channel = kk*8 + sub)
        int lane = t & 31, sub = t >> 5;
        float mx = -3.4e38f;
        for (int kk = lane; kk < 125; kk += 32)
            mx = fmaxf(mx, sLog[kk * 8 + sub]);
        #pragma unroll
        for (int s = 16; s > 0; s >>= 1) mx = fmaxf(mx, __shfl_xor(mx, s, 32));
        float sum = 0.f;
        for (int kk = lane; kk < 125; kk += 32) {
            float e = __expf(sLog[kk * 8 + sub] - mx);
            sW[sub * 125 + kk] = e;
            sum += e;
        }
        #pragma unroll
        for (int s = 16; s > 0; s >>= 1) sum += __shfl_xor(sum, s, 32);
        float inv = 1.f / sum;
        for (int kk = lane; kk < 125; kk += 32) sW[sub * 125 + kk] *= inv;
    }
    __syncthreads();

    int c = t & 31, sub = t >> 5;
    float acc = 0.f;
    for (int tap = 0; tap < 125; ++tap)
        acc = fmaf(sW[sub * 125 + tap], sX[c * 125 + tap], acc);

    int H = 2 * h0 + (sub >> 2);
    int W = 2 * w0 + ((sub >> 1) & 1);
    int D = 2 * d0 + (sub & 1);
    out[(((size_t)(b * 32 + c) * 64 + H) * 64 + W) * 64 + D] = acc;
}

// ---------------- launcher ---------------------------------------------------
extern "C" void kernel_launch(void* const* d_in, const int* in_sizes, int n_in,
                              void* d_out, int out_size, void* d_ws, size_t ws_size,
                              hipStream_t stream)
{
    const float* X  = (const float*)d_in[0];
    const float* wc = (const float*)d_in[1];
    const float* g1 = (const float*)d_in[2];
    const float* b1 = (const float*)d_in[3];
    const float* m1 = (const float*)d_in[4];
    const float* v1 = (const float*)d_in[5];
    const float* we = (const float*)d_in[6];
    const float* g2 = (const float*)d_in[7];
    const float* b2 = (const float*)d_in[8];
    const float* m2 = (const float*)d_in[9];
    const float* v2 = (const float*)d_in[10];
    float* out = (float*)d_out;

    char* ws = (char*)d_ws;
    __hip_bfloat16* W1hi = (__hip_bfloat16*)ws;                      //  8,388,608 B
    __hip_bfloat16* W1lo = (__hip_bfloat16*)(ws +  8388608);         //  8,388,608 B
    __hip_bfloat16* Aphi = (__hip_bfloat16*)(ws + 16777216);         //  3,538,944 B
    __hip_bfloat16* Aplo = (__hip_bfloat16*)(ws + 20316160);         //  3,538,944 B
    float* out2 = (float*)(ws + 23855104);                           // 264,241,152 B

    k_pointwise<<<NTOT / 256, 256, 0, stream>>>(X, wc, g1, b1, m1, v1, W1hi, W1lo);
    k_pack_w<<<(1024 * KTOT) / 256, 256, 0, stream>>>(we, Aphi, Aplo);
    k_conv_gemm<<<dim3(NTOT / 64, 8), 256, 0, stream>>>(Aphi, Aplo, W1hi, W1lo,
                                                        g2, b2, m2, v2, out2);
    k_carafe<<<NTOT, 256, 0, stream>>>(out2, X, out);
}
